// SpatialTemporalEnhancedLayer_90159953477914
// MI455X (gfx1250) — compile-verified, matrix operand order checked by reading, not run on hardware
//
#include <hip/hip_runtime.h>
#include <hip/hip_bf16.h>

// ---------------------------------------------------------------------------
// MI455X (gfx1250) implementation of SpatialTemporalEnhancedLayer forward.
//
// Roofline: ~95 GFLOP vs ~1 GB HBM traffic -> memory bound at 23.3 TB/s.
// All GEMMs use v_wmma_f32_16x16x32_f16 (f16 in, f32 acc), fp32->f16 converted
// in registers. Inner loops are unconditional (no acc phis), addresses are
// hoisted, and edge masking runs only on wave-uniform edge tiles.
//
// Auto-correlation: corr is only reduced over (H,E), so
//   mean_value[b,l] = (1/128) * sum_j S[b,j,(j-l)%L],  S[b] = q[b] @ k[b]^T
// (batched WMMA Gram matrix); aggregation = 5 weighted circular rolls of v.
// ---------------------------------------------------------------------------

#define B_  512
#define L_  200
#define D_  128
#define TOPK 5
#define ML  (B_ * L_)           // 102400 rows, multiple of 64
#define MT  4                   // M-subtiles (16 rows each) per wave

typedef __attribute__((ext_vector_type(16))) _Float16 v16h;
typedef __attribute__((ext_vector_type(8)))  float    v8f;

// Two contiguous 8-float chunks (p, p+16) -> 16-bit WMMA operand fragment
// (ISA 7.12.2: VGPR j holds K-pair {kb+2j,kb+2j+1} j<4, {kb+16+2(j-4),..} j>=4).
__device__ __forceinline__ v16h frag_from_ptr(const float* __restrict__ p) {
  float4 a = *(const float4*)(p);
  float4 b = *(const float4*)(p + 4);
  float4 c = *(const float4*)(p + 16);
  float4 d = *(const float4*)(p + 20);
  v16h o;
  o[0]  = (_Float16)a.x; o[1]  = (_Float16)a.y; o[2]  = (_Float16)a.z; o[3]  = (_Float16)a.w;
  o[4]  = (_Float16)b.x; o[5]  = (_Float16)b.y; o[6]  = (_Float16)b.z; o[7]  = (_Float16)b.w;
  o[8]  = (_Float16)c.x; o[9]  = (_Float16)c.y; o[10] = (_Float16)c.z; o[11] = (_Float16)c.w;
  o[12] = (_Float16)d.x; o[13] = (_Float16)d.y; o[14] = (_Float16)d.z; o[15] = (_Float16)d.w;
  return o;
}

// Masked variant: same loads (row pointer is pre-clamped, K overread is
// harmless), then zero out-of-range elements. kabs = k0 + kb0 (absolute K).
__device__ __forceinline__ v16h frag_masked(const float* __restrict__ p,
                                            int kabs, int kmax, bool rok) {
  v16h o = frag_from_ptr(p);
#pragma unroll
  for (int i = 0; i < 8; ++i) {
    if (!(rok && (kabs + i)      < kmax)) o[i]     = (_Float16)0.0f;
    if (!(rok && (kabs + 16 + i) < kmax)) o[8 + i] = (_Float16)0.0f;
  }
  return o;
}

// Column fragment from a [K, 128] row-major matrix: p = Bm + col, elements at
// absolute K = kabs + {0..7, 16..23}, stride 128 floats (immediate offsets).
__device__ __forceinline__ v16h colfrag(const float* __restrict__ p, int kabs) {
  const float* q = p + (long)kabs * D_;
  v16h o;
#pragma unroll
  for (int i = 0; i < 8; ++i) {
    o[i]     = (_Float16)q[i * D_];
    o[8 + i] = (_Float16)q[(16 + i) * D_];
  }
  return o;
}

__device__ __forceinline__ v16h colfrag_masked(const float* __restrict__ p,
                                               int kabs, int kmax, bool cok) {
  v16h o;
#pragma unroll
  for (int i = 0; i < 8; ++i) {
    int ka = kabs + i, kc = kabs + 16 + i;
    int ca = (ka < kmax) ? ka : (kmax - 1);
    int cb = (kc < kmax) ? kc : (kmax - 1);
    float va = p[(long)ca * D_];
    float vb = p[(long)cb * D_];
    o[i]     = (cok && ka < kmax) ? (_Float16)va : (_Float16)0.0f;
    o[8 + i] = (cok && kc < kmax) ? (_Float16)vb : (_Float16)0.0f;
  }
  return o;
}

// --- Generic batched WMMA GEMM ---------------------------------------------
// C[b] = A[b] (MxK, row-major) @ B[b] (+ bias[n], + optional relu)
//   B_ROWS=true : Bm stored [N,K] row-major -> C = A @ Bm^T   (linears, Gram)
//   B_ROWS=false: Bm stored [K,128] row-major -> C = A @ Bm   (adj@E, @weight)
// One wave per block; 64x16 output strip (MT subtiles), one B fragment per
// k-step feeds MT WMMAs. All loop-body work is unconditional; OOB subtiles
// accumulate zeros (clamped rows + masks) and skip only the store.
template <bool B_ROWS>
__global__ void wmma_gemm(const float* __restrict__ A,
                          const float* __restrict__ Bm,
                          const float* __restrict__ bias,
                          float* __restrict__ C,
                          int M, int N, int K,
                          long sA, long sB, long sC, int relu) {
  const float* Ab = A  + (long)blockIdx.z * sA;
  const float* Bb = Bm + (long)blockIdx.z * sB;
  float*       Cb = C  + (long)blockIdx.z * sC;
  int m0 = blockIdx.x * (16 * MT);
  int n0 = blockIdx.y << 4;

  int t   = threadIdx.x & 31;
  int lr  = t & 15;            // row (A) / col (B) within tile
  int kb0 = (t >> 4) << 3;     // 0 or 8: lane K-offset

  // Hoisted per-subtile A row pointers (clamped rows keep 16B alignment:
  // K is a multiple of 8 floats and kb0 is 0/8).
  const float* pA[MT];
  bool rok[MT];
#pragma unroll
  for (int i = 0; i < MT; ++i) {
    int r  = m0 + 16 * i + lr;
    int rc = (r < M) ? r : (M - 1);
    pA[i]  = Ab + (long)rc * K + kb0;
    rok[i] = (r < M);
  }
  // Hoisted B pointer.
  int  cB  = n0 + lr;
  bool cok = (cB < N);
  int  cBc = cok ? cB : (N - 1);
  const float* pBr = Bb + (long)cBc * K + kb0;   // B_ROWS ([N,K])
  const float* pBc = Bb + cBc;                   // cols   ([K,128])

  v8f acc[MT] = {};
  const int Kfull = K & ~31;
  const bool interior = (m0 + 16 * MT <= M) && (n0 + 16 <= N);

  if (interior) {
    for (int k0 = 0; k0 < Kfull; k0 += 32) {
      v16h bv = B_ROWS ? frag_from_ptr(pBr + k0) : colfrag(pBc, k0 + kb0);
#pragma unroll
      for (int i = 0; i < MT; ++i) {
        v16h av = frag_from_ptr(pA[i] + k0);
        acc[i] = __builtin_amdgcn_wmma_f32_16x16x32_f16(
            false, av, false, bv, (short)0, acc[i], false, false);
      }
    }
    if (Kfull < K) {                       // K tail (adj GEMM: K=200)
      int k0 = Kfull, ka = k0 + kb0;
      v16h bv = B_ROWS ? frag_masked(pBr + k0, ka, K, true)
                       : colfrag_masked(pBc, ka, K, true);
#pragma unroll
      for (int i = 0; i < MT; ++i) {
        v16h av = frag_masked(pA[i] + k0, ka, K, true);
        acc[i] = __builtin_amdgcn_wmma_f32_16x16x32_f16(
            false, av, false, bv, (short)0, acc[i], false, false);
      }
    }
  } else {                                 // M/N edge blocks: fully masked
    for (int k0 = 0; k0 < K; k0 += 32) {
      int ka = k0 + kb0;
      v16h bv = B_ROWS ? frag_masked(pBr + k0, ka, K, cok)
                       : colfrag_masked(pBc, ka, K, cok);
#pragma unroll
      for (int i = 0; i < MT; ++i) {
        v16h av = frag_masked(pA[i] + k0, ka, K, rok[i]);
        acc[i] = __builtin_amdgcn_wmma_f32_16x16x32_f16(
            false, av, false, bv, (short)0, acc[i], false, false);
      }
    }
  }

  // D layout: lane t -> n = n0 + (t&15); VGPR j -> m = mi + j + 8*(t>=16)
  int n = n0 + lr;
  if (n < N) {
    float bvv = bias ? bias[n] : 0.0f;
    int mhi = (t >> 4) << 3;
#pragma unroll
    for (int i = 0; i < MT; ++i) {
      int mi = m0 + 16 * i;
#pragma unroll
      for (int j = 0; j < 8; ++j) {
        int m = mi + j + mhi;
        if (m < M) {
          float val = acc[i][j] + bvv;
          if (relu) val = fmaxf(val, 0.0f);
          Cb[(long)m * N + n] = val;
        }
      }
    }
  }
}

// --- Elementwise / reduction kernels ---------------------------------------
__global__ void gather_nodes(const float* __restrict__ tab,
                             const int* __restrict__ idx,
                             float* __restrict__ out) {
  int i = blockIdx.x, c = threadIdx.x;                  // i over B*L
  out[(long)i * D_ + c] = tab[(long)idx[i] * D_ + c];
}

__global__ void build_totals(const float* __restrict__ E,
                             const float* __restrict__ g1,
                             const float* __restrict__ g2,
                             const int* __restrict__ seqs_cate,
                             const int* __restrict__ masks,
                             const int* __restrict__ ts_poi,
                             const int* __restrict__ ts_cate,
                             const float* __restrict__ cate_embeds,
                             const float* __restrict__ pos_emb,
                             const float* __restrict__ pos_emb_cate,
                             const float* __restrict__ time_emb,
                             const float* __restrict__ time_emb_cate,
                             float* __restrict__ total,
                             float* __restrict__ cate_total) {
  int i = blockIdx.x, c = threadIdx.x;                  // i = b*L + l
  int l   = i % L_;
  int pos = (l + 1) * masks[i];                         // row 0 of pos_emb is 0
  int tp  = ts_poi[i]  - (10000 + 100000);              // NUM_USERS+NUM_POIS
  int tc  = ts_cate[i] - (10000 + 500);                 // NUM_USERS+NUM_CATE
  long o = (long)i * D_ + c;
  total[o] = E[o] + pos_emb[(long)pos * D_ + c] + g1[o] + g2[o] +
             time_emb[(long)tp * D_ + c];
  cate_total[o] = cate_embeds[(long)seqs_cate[i] * D_ + c] +
                  pos_emb_cate[(long)pos * D_ + c] +
                  time_emb_cate[(long)tc * D_ + c];
}

__global__ void zero_f(float* __restrict__ p, int n) {
  int i = blockIdx.x * blockDim.x + threadIdx.x;
  if (i < n) p[i] = 0.0f;
}

// mean_value[b,l] = (1/128) * sum_j S[b, j, (j-l) mod L]; gmean[l] += mv/B
__global__ void mean_corr(const float* __restrict__ S,
                          float* __restrict__ mean_value,
                          float* __restrict__ gmean) {
  int i = blockIdx.x * blockDim.x + threadIdx.x;        // over B*L
  if (i >= ML) return;
  int b = i / L_, l = i % L_;
  const float* Sb = S + (long)b * L_ * L_;
  float s = 0.0f;
  for (int j = 0; j < L_; ++j) {
    int jj = j - l; if (jj < 0) jj += L_;
    s += Sb[j * L_ + jj];
  }
  s *= (1.0f / (float)D_);
  mean_value[i] = s;
  atomicAdd(&gmean[l], s * (1.0f / (float)B_));
}

// Global top-5 delays of gmean[L]; first-index tie-break like jax.lax.top_k.
__global__ void topk5(const float* __restrict__ gmean, int* __restrict__ idx) {
  if (threadIdx.x != 0 || blockIdx.x != 0) return;
  float g[L_];
  for (int i = 0; i < L_; ++i) g[i] = gmean[i];
  for (int k = 0; k < TOPK; ++k) {
    int best = 0; float bv = -3.0e38f;
    for (int i = 0; i < L_; ++i) if (g[i] > bv) { bv = g[i]; best = i; }
    idx[k] = best; g[best] = -3.0e38f;
  }
}

// agg[b,l,c] = sum_k softmax_k(mean_value[b,idx[k]]) * v[b,(l+idx[k])%L,c]
__global__ void softmax_agg(const float* __restrict__ mean_value,
                            const int* __restrict__ idx,
                            const float* __restrict__ v,
                            float* __restrict__ agg) {
  int i = blockIdx.x, c = threadIdx.x;                  // i = b*L + l
  int b = i / L_, l = i % L_;
  int d[TOPK]; float w[TOPK];
  float mx = -3.0e38f;
#pragma unroll
  for (int k = 0; k < TOPK; ++k) {
    d[k] = idx[k];
    w[k] = mean_value[b * L_ + d[k]];
    mx = fmaxf(mx, w[k]);
  }
  float ssum = 0.0f;
#pragma unroll
  for (int k = 0; k < TOPK; ++k) { w[k] = expf(w[k] - mx); ssum += w[k]; }
  float inv = 1.0f / ssum;
  const float* vb = v + (long)b * L_ * D_;
  float acc = 0.0f;
#pragma unroll
  for (int k = 0; k < TOPK; ++k) {
    int ll = l + d[k]; if (ll >= L_) ll -= L_;
    acc += w[k] * inv * vb[(long)ll * D_ + c];
  }
  agg[(long)i * D_ + c] = acc;
}

__global__ void add2(const float* __restrict__ a, const float* __restrict__ b,
                     float* __restrict__ o, long n) {
  long i = (long)blockIdx.x * blockDim.x + threadIdx.x;
  if (i < n) o[i] = a[i] + b[i];
}

// ---------------------------------------------------------------------------
extern "C" void kernel_launch(void* const* d_in, const int* in_sizes, int n_in,
                              void* d_out, int out_size, void* d_ws, size_t ws_size,
                              hipStream_t stream) {
  (void)in_sizes; (void)n_in; (void)out_size; (void)ws_size;

  // Input order = setup_inputs() dict insertion order (params flattened last,
  // nested dicts in insertion order: W before b, q/k/v/o per AC layer).
  const float* nodes_embeds = (const float*)d_in[2];
  const float* cate_embeds  = (const float*)d_in[3];
  const int*   seqs         = (const int*)  d_in[4];
  const int*   seqs_cate    = (const int*)  d_in[5];
  const int*   masks        = (const int*)  d_in[6];
  const float* geo_adjs     = (const float*)d_in[7];
  const float* time_adjs    = (const float*)d_in[8];
  const int*   ts_poi       = (const int*)  d_in[10];
  const int*   ts_cate      = (const int*)  d_in[11];
  const float* pos_emb      = (const float*)d_in[12];
  const float* pos_emb_cate = (const float*)d_in[13];
  const float* time_emb     = (const float*)d_in[14];
  const float* time_emb_cate= (const float*)d_in[15];
  const float* Wgeo  = (const float*)d_in[16];
  const float* bgeo  = (const float*)d_in[17];
  const float* Wtime = (const float*)d_in[18];
  const float* btime = (const float*)d_in[19];
  const float* weight= (const float*)d_in[20];
  const int AC_BASE[4] = {21, 29, 37, 45};  // ac, ac_cate, ac_cross, ac_cate_cross

  // Workspace layout (fp32 slots of B*L*D)
  const size_t SZ = (size_t)ML * D_;
  float* W = (float*)d_ws;
  float* total      = W + 0 * SZ;
  float* cate_total = W + 1 * SZ;
  float* qb   = W + 2 * SZ;
  float* kb   = W + 3 * SZ;
  float* vb   = W + 4 * SZ;
  float* agg  = W + 5 * SZ;
  float* mha  = W + 6 * SZ;
  float* cmha = W + 7 * SZ;
  float* crss = W + 8 * SZ;
  float* ccrs = W + 9 * SZ;
  float* E    = W + 10 * SZ;   // dead after build_totals; S overlays 10..11.6
  float* t1   = W + 11 * SZ;
  float* t2   = W + 12 * SZ;
  float* g1   = W + 13 * SZ;
  float* g2   = W + 14 * SZ;
  float* S    = W + 10 * SZ;   // [B, L, L] = 20.48M floats, reuses E/t1 region
  float* mean_value = W + 15 * SZ;          // [B*L]
  float* gmean      = mean_value + ML;      // [L] (padded)
  int*   idxbuf     = (int*)(gmean + 256);  // [TOPK]

  const dim3 blk(32);
  const int  MR = 16 * MT;                              // rows per block
  const dim3 gLin((ML + MR - 1) / MR, D_ / 16, 1);      // 1600 x 8
  const dim3 gAdj((L_ + MR - 1) / MR, D_ / 16, B_);     // 4 x 8 x 512
  const dim3 gGram((L_ + MR - 1) / MR, (L_ + 15) / 16, B_); // 4 x 13 x 512
  const long sLD = (long)L_ * D_, sLL = (long)L_ * L_;

  // 1) gather POI/user embeddings
  gather_nodes<<<ML, D_, 0, stream>>>(nodes_embeds, seqs, E);

  // 2) adj @ E (batched, B operand column-loaded from [K=L, N=D])
  wmma_gemm<false><<<gAdj, blk, 0, stream>>>(geo_adjs,  E, nullptr, t1,
                                             L_, D_, L_, sLL, sLD, sLD, 0);
  wmma_gemm<false><<<gAdj, blk, 0, stream>>>(time_adjs, E, nullptr, t2,
                                             L_, D_, L_, sLL, sLD, sLD, 0);

  // 3) relu(linear) for geo / time
  wmma_gemm<true><<<gLin, blk, 0, stream>>>(t1, Wgeo,  bgeo,  g1,
                                            ML, D_, D_, 0, 0, 0, 1);
  wmma_gemm<true><<<gLin, blk, 0, stream>>>(t2, Wtime, btime, g2,
                                            ML, D_, D_, 0, 0, 0, 1);

  // 4) total / cate_total
  build_totals<<<ML, D_, 0, stream>>>(E, g1, g2, seqs_cate, masks, ts_poi,
                                      ts_cate, cate_embeds, pos_emb,
                                      pos_emb_cate, time_emb, time_emb_cate,
                                      total, cate_total);

  // 5) auto-correlation attention layer
  auto run_ac = [&](const float* Qin, const float* Kin, const float* Vin,
                    int pb, float* Out) {
    const float* Wq = (const float*)d_in[pb + 0]; const float* bq = (const float*)d_in[pb + 1];
    const float* Wk = (const float*)d_in[pb + 2]; const float* bk2= (const float*)d_in[pb + 3];
    const float* Wv = (const float*)d_in[pb + 4]; const float* bv2= (const float*)d_in[pb + 5];
    const float* Wo = (const float*)d_in[pb + 6]; const float* bo = (const float*)d_in[pb + 7];
    wmma_gemm<true><<<gLin, blk, 0, stream>>>(Qin, Wq, bq,  qb, ML, D_, D_, 0, 0, 0, 0);
    wmma_gemm<true><<<gLin, blk, 0, stream>>>(Kin, Wk, bk2, kb, ML, D_, D_, 0, 0, 0, 0);
    wmma_gemm<true><<<gLin, blk, 0, stream>>>(Vin, Wv, bv2, vb, ML, D_, D_, 0, 0, 0, 0);
    // Gram S[b] = q[b] @ k[b]^T   (circular cross-correlation source)
    wmma_gemm<true><<<gGram, blk, 0, stream>>>(qb, kb, nullptr, S,
                                               L_, L_, D_, sLD, sLD, sLL, 0);
    zero_f<<<1, 256, 0, stream>>>(gmean, 256);
    mean_corr<<<(ML + 255) / 256, 256, 0, stream>>>(S, mean_value, gmean);
    topk5<<<1, 32, 0, stream>>>(gmean, idxbuf);
    softmax_agg<<<ML, D_, 0, stream>>>(mean_value, idxbuf, vb, agg);
    wmma_gemm<true><<<gLin, blk, 0, stream>>>(agg, Wo, bo, Out, ML, D_, D_, 0, 0, 0, 0);
  };

  run_ac(total,      total,      total,      AC_BASE[0], mha);
  run_ac(cate_total, cate_total, cate_total, AC_BASE[1], cmha);
  run_ac(mha,        cmha,       cmha,       AC_BASE[2], crss);
  run_ac(cmha,       mha,        mha,        AC_BASE[3], ccrs);

  // 6) (cross + cate_cross) @ weight  -> d_out  (weight is [K,128], col loads)
  add2<<<(int)((SZ + 255) / 256), 256, 0, stream>>>(crss, ccrs, total, (long)SZ);
  wmma_gemm<false><<<gLin, blk, 0, stream>>>(total, weight, nullptr,
                                             (float*)d_out, ML, D_, D_,
                                             0, 0, 0, 0);
}
